// MultiHeadAttentionLayer_16329465660110
// MI455X (gfx1250) — matrix operand order repaired and checked by reading, not compiled, not run on hardware
//
#include <hip/hip_runtime.h>

typedef __attribute__((ext_vector_type(16))) _Float16 v16h;
typedef __attribute__((ext_vector_type(8)))  _Float16 v8h;
typedef __attribute__((ext_vector_type(8)))  float    v8f;
typedef __attribute__((ext_vector_type(4)))  float    v4f;
typedef __attribute__((ext_vector_type(4)))  unsigned v4u;
typedef __attribute__((ext_vector_type(8)))  int      v8i_t;
typedef __attribute__((ext_vector_type(4)))  int      v4i_t;

#define HID   1024
#define HEADS 16
#define HDIM  64
#define SEQ   2048
#define BATCH 2
#define MTOT  (BATCH*SEQ)   // 4096

// ---------------------------------------------------------------------------
// TDM: issue a 2D f16 tile load Global->LDS (Tensor Data Mover, §8 D# format).
// tw = elements per row, th = rows, rstride = row stride in elements.
// Tensor dims == tile dims (tile always fully in-bounds). LDS fill is
// contiguous row-major. Tracked by TENSORcnt.
// ---------------------------------------------------------------------------
__device__ __forceinline__ void tdm_load_2d(unsigned lds_off, const void* gptr,
                                            unsigned tw, unsigned th,
                                            unsigned rstride) {
    unsigned long long ga = (unsigned long long)(size_t)gptr;
    v4u g0;
    g0[0] = 1u;                                   // count=1 (valid user D#)
    g0[1] = lds_off;                              // lds_addr (bytes)
    g0[2] = (unsigned)ga;                         // global_addr[31:0]
    g0[3] = (unsigned)((ga >> 32) & 0x01ffffffull) | (2u << 30); // addr[56:32] | type=2
    v8i_t g1;
    g1[0] = (int)(1u << 16);                      // data_size=1 (2 bytes)
    g1[1] = (int)(tw << 16);                      // tensor_dim0[15:0]
    g1[2] = (int)((tw >> 16) | (th << 16));       // tensor_dim0[31:16] | tensor_dim1[15:0]
    g1[3] = (int)((th >> 16) | (tw << 16));       // tensor_dim1[31:16] | tile_dim0
    g1[4] = (int)th;                              // tile_dim1 (tile_dim2=0)
    g1[5] = (int)rstride;                         // tensor_dim0_stride[31:0]
    g1[6] = 0;
    g1[7] = 0;
    v4i_t gz4 = {};                               // groups 2/3: dims 2..4 unused
    v8i_t gz8 = {};                               // trailing group (clang-23 6-arg form)
    __builtin_amdgcn_tensor_load_to_lds(g0, g1, gz4, gz4, gz8, 0);
}

// Build a 16x32 f16 A-fragment (v16h) from two contiguous 8-half runs.
// Per CDNA5 ISA 7.12.2: lane<16 holds row=lane, K = {h8+0..7, h8+16..23}
__device__ __forceinline__ v16h frag_from2(const _Float16* p0, const _Float16* p1) {
    v8h a = *(const v8h*)p0;
    v8h b = *(const v8h*)p1;
    v16h r;
#pragma unroll
    for (int i = 0; i < 8; ++i) { r[i] = a[i]; r[i + 8] = b[i]; }
    return r;
}

__device__ __forceinline__ float red_max16(float x) {
#pragma unroll
    for (int m = 1; m < 16; m <<= 1) x = fmaxf(x, __shfl_xor(x, m, 32));
    return x;
}
__device__ __forceinline__ float red_sum16(float x) {
#pragma unroll
    for (int m = 1; m < 16; m <<= 1) x += __shfl_xor(x, m, 32);
    return x;
}

// ---------------------------------------------------------------------------
// Kernel 1: convert 4 fp32 weight matrices [K=1024][N=1024] into f16 WT[n][k]
// (transposed) so GEMM B-fragments become pairs of contiguous 16B runs.
// ---------------------------------------------------------------------------
__global__ void __launch_bounds__(256) k_wT(const float* w0, const float* w1,
                                            const float* w2, const float* w3,
                                            _Float16* o0, _Float16* o1,
                                            _Float16* o2, _Float16* o3) {
    const float* W;
    _Float16*    O;
    switch (blockIdx.z) {
        case 0: W = w0; O = o0; break;
        case 1: W = w1; O = o1; break;
        case 2: W = w2; O = o2; break;
        default: W = w3; O = o3; break;
    }
    __shared__ float tile[64][65];
    const int t  = threadIdx.x;
    const int kb = blockIdx.y * 64, nb = blockIdx.x * 64;
    const int r  = t >> 2, c0 = (t & 3) * 16;
#pragma unroll
    for (int i = 0; i < 16; i += 4) {
        v4f v = *(const v4f*)(W + (size_t)(kb + r) * HID + nb + c0 + i);
#pragma unroll
        for (int j = 0; j < 4; ++j) tile[r][c0 + i + j] = v[j];
    }
    __syncthreads();
    v8h h0, h1;
#pragma unroll
    for (int i = 0; i < 8; ++i) {
        h0[i] = (_Float16)tile[c0 + i][r];
        h1[i] = (_Float16)tile[c0 + 8 + i][r];
    }
    _Float16* dst = O + (size_t)(nb + r) * HID + kb + c0;
    *(v8h*)dst       = h0;
    *(v8h*)(dst + 8) = h1;
}

// ---------------------------------------------------------------------------
// Kernel 2: WMMA GEMM  C[4096x1024] = A[4096x1024] * W + bias
//   AF16: A fp32 (0) or f16 (1), row-major.  WT: f16 transposed weights WT[n][k].
//   OMODE 0: f16 out [B,H,L,D] | 1: f16 out [B,H,D,L] (packed b128) | 2: f32 [M][N]
// Block: 256 thr = 8 waves; tile 64(M) x 128(N), BK=64; wave tile 32x32.
// B tiles are TDM double-buffered (wave 0 drives the DMA), A vector-staged.
// ---------------------------------------------------------------------------
template <int AF16, int OMODE>
__global__ void __launch_bounds__(256) k_gemm(const void* Aptr,
                                              const _Float16* WT, const float* bias,
                                              void* outp, float scale) {
    __shared__ _Float16 sA[64 * 64];
    __shared__ _Float16 sB[2][128 * 64];
    const int t    = threadIdx.x;
    const int Nb   = blockIdx.x * 128, Mb = blockIdx.y * 64;
    const int lane = t & 31, wid = t >> 5;
    const int r15  = lane & 15, h8 = (lane >> 4) * 8, h16 = (lane >> 4) * 16;
    const int wm   = wid & 1, wn = wid >> 1;

    v8f acc[2][2];
    v8f zf = {};
#pragma unroll
    for (int i = 0; i < 2; ++i)
#pragma unroll
        for (int j = 0; j < 2; ++j) acc[i][j] = zf;

    const int a_r = t >> 2, a_k = (t & 3) * 16;

    // prologue: DMA first B tile
    if (wid == 0)
        tdm_load_2d((unsigned)(size_t)(void*)&sB[0][0],
                    WT + (size_t)Nb * HID, 64, 128, HID);

    int buf = 0;
    for (int k0 = 0; k0 < HID; k0 += 64) {
        // stage A tile (64x64 f16), all threads
        if (AF16) {
            const _Float16* src = (const _Float16*)Aptr + (size_t)(Mb + a_r) * HID + k0 + a_k;
            if (k0 + 64 < HID) __builtin_prefetch(src + 64, 0, 1);
            *(v8h*)(sA + a_r * 64 + a_k)     = *(const v8h*)src;
            *(v8h*)(sA + a_r * 64 + a_k + 8) = *(const v8h*)(src + 8);
        } else {
            const float* src = (const float*)Aptr + (size_t)(Mb + a_r) * HID + k0 + a_k;
            if (k0 + 64 < HID) __builtin_prefetch(src + 64, 0, 1);
            v4f f0 = *(const v4f*)(src);
            v4f f1 = *(const v4f*)(src + 4);
            v4f f2 = *(const v4f*)(src + 8);
            v4f f3 = *(const v4f*)(src + 12);
            v8h h0, h1;
#pragma unroll
            for (int i = 0; i < 4; ++i) {
                h0[i] = (_Float16)f0[i]; h0[4 + i] = (_Float16)f1[i];
                h1[i] = (_Float16)f2[i]; h1[4 + i] = (_Float16)f3[i];
            }
            *(v8h*)(sA + a_r * 64 + a_k)     = h0;
            *(v8h*)(sA + a_r * 64 + a_k + 8) = h1;
        }
        // wave 0: prefetch next B tile via TDM, ensure current tile landed
        if (wid == 0) {
            if (k0 + 64 < HID) {
                tdm_load_2d((unsigned)(size_t)(void*)&sB[buf ^ 1][0],
                            WT + (size_t)Nb * HID + (k0 + 64), 64, 128, HID);
                __builtin_amdgcn_s_wait_tensorcnt(1);
            } else {
                __builtin_amdgcn_s_wait_tensorcnt(0);
            }
        }
        __syncthreads();
        const _Float16* bbase = &sB[buf][0];
#pragma unroll
        for (int ks = 0; ks < 2; ++ks) {
            v16h aF[2], bF[2];
#pragma unroll
            for (int ms = 0; ms < 2; ++ms) {
                const _Float16* p = sA + (wm * 32 + ms * 16 + r15) * 64 + ks * 32 + h8;
                aF[ms] = frag_from2(p, p + 16);
            }
#pragma unroll
            for (int ns = 0; ns < 2; ++ns) {
                const _Float16* p = bbase + (wn * 32 + ns * 16 + r15) * 64 + ks * 32 + h8;
                bF[ns] = frag_from2(p, p + 16);
            }
#pragma unroll
            for (int ms = 0; ms < 2; ++ms)
#pragma unroll
                for (int ns = 0; ns < 2; ++ns)
                    acc[ms][ns] = __builtin_amdgcn_wmma_f32_16x16x32_f16(
                        false, aF[ms], false, bF[ns], (short)0, acc[ms][ns], false, false);
        }
        __syncthreads();
        buf ^= 1;
    }

    // epilogue
#pragma unroll
    for (int ms = 0; ms < 2; ++ms) {
#pragma unroll
        for (int ns = 0; ns < 2; ++ns) {
            const int   n     = Nb + wn * 32 + ns * 16 + r15;
            const float bval  = bias[n];
            const int   mbase = Mb + wm * 32 + ms * 16 + h8;  // 8 consecutive rows
            if (OMODE == 1) {
                const int b = mbase >> 11, l = mbase & (SEQ - 1);
                const int h = n >> 6, d = n & 63;
                v8h pk;
#pragma unroll
                for (int v = 0; v < 8; ++v) pk[v] = (_Float16)((acc[ms][ns][v] + bval) * scale);
                *(v8h*)((_Float16*)outp + (((size_t)(b * HEADS + h) * HDIM + d) * SEQ + l)) = pk;
            } else if (OMODE == 0) {
#pragma unroll
                for (int v = 0; v < 8; ++v) {
                    const int m = mbase + v;
                    const int b = m >> 11, l = m & (SEQ - 1);
                    const int h = n >> 6, d = n & 63;
                    ((_Float16*)outp)[((size_t)(b * HEADS + h) * SEQ + l) * HDIM + d] =
                        (_Float16)((acc[ms][ns][v] + bval) * scale);
                }
            } else {
#pragma unroll
                for (int v = 0; v < 8; ++v) {
                    const int m = mbase + v;
                    ((float*)outp)[(size_t)m * HID + n] = (acc[ms][ns][v] + bval) * scale;
                }
            }
        }
    }
}

// ---------------------------------------------------------------------------
// Kernel 3: srel[row][j] = qh_scaled[row,:] . rel_k[j,:]   (includes 1/8 scale)
// ---------------------------------------------------------------------------
__global__ void __launch_bounds__(256) k_srel(const _Float16* qh, const float* rel_k,
                                              float* srel) {
    const int idx = blockIdx.x * 256 + threadIdx.x;
    if (idx >= BATCH * HEADS * SEQ * 5) return;
    const int j   = idx % 5;
    const int row = idx / 5;
    const _Float16* q  = qh + (size_t)row * HDIM;
    const float*    rk = rel_k + j * HDIM;
    float acc = 0.f;
#pragma unroll
    for (int d = 0; d < HDIM; ++d) acc += (float)q[d] * rk[d];
    srel[idx] = acc;
}

// ---------------------------------------------------------------------------
// Kernel 4: flash attention with relative embeddings.
// Grid (L/64, HEADS, B), 128 threads = 4 waves; wave owns 16 query rows.
// qh: [B,H,L,D] f16 (scaled 1/8); kh: [B,H,L,D] f16; vhT: [B,H,D,L] f16.
// K / V^T tiles are TDM double-buffered (wave 0 drives the DMA).
// ctx out: [B,L,HID] f16.
// ---------------------------------------------------------------------------
__global__ void __launch_bounds__(128) k_attn(const _Float16* qh, const _Float16* kh,
                                              const _Float16* vhT, const float* srel_all,
                                              const float* rel_v, _Float16* ctx) {
    __shared__ _Float16 sK[2][64 * 64];   // K tile   [key][d]
    __shared__ _Float16 sVT[2][64 * 64];  // V^T tile [d][key]
    __shared__ _Float16 sP[4 * 16 * 64];  // per-wave P staging [row][key]
    __shared__ float    sRel[5 * 64];

    const int t    = threadIdx.x;
    const int lane = t & 31, w = t >> 5;
    const int r15  = lane & 15;
    const int h8   = (lane >> 4) * 8, h16 = (lane >> 4) * 16;
    const int qb   = blockIdx.x * 64;
    const int bh   = blockIdx.z * HEADS + blockIdx.y;

    const _Float16* khb = kh + (size_t)bh * SEQ * HDIM;
    const _Float16* vtb = vhT + (size_t)bh * HDIM * SEQ;

    // prologue: DMA first K/V^T tiles
    if (w == 0) {
        tdm_load_2d((unsigned)(size_t)(void*)&sK[0][0],  khb, 64, 64, HDIM);
        tdm_load_2d((unsigned)(size_t)(void*)&sVT[0][0], vtb, 64, 64, SEQ);
    }

    for (int i = t; i < 5 * 64; i += 128) sRel[i] = rel_v[i];

    // resident Q fragments (rows qb+w*16 .. +15, D=64 -> 2 k-steps)
    v16h aQ[2];
    {
        const _Float16* qp = qh + ((size_t)bh * SEQ + qb + w * 16 + r15) * HDIM;
#pragma unroll
        for (int ks = 0; ks < 2; ++ks) {
            const _Float16* p = qp + ks * 32 + h8;
            aQ[ks] = frag_from2(p, p + 16);
        }
    }
    // per-row (C-layout row = v + h8) relative-key scores
    float srl[8][5];
#pragma unroll
    for (int v = 0; v < 8; ++v) {
        const int    row = qb + w * 16 + v + h8;
        const float* sp  = srel_all + ((size_t)bh * SEQ + row) * 5;
#pragma unroll
        for (int j = 0; j < 5; ++j) srl[v][j] = sp[j];
    }

    float m[8], l[8], g1[8], g2[8], g3[8], g4[8];
    v8f   o[4];
    v8f   zf = {};
#pragma unroll
    for (int v = 0; v < 8; ++v) {
        m[v] = -1e30f; l[v] = 0.f;
        g1[v] = 0.f; g2[v] = 0.f; g3[v] = 0.f; g4[v] = 0.f;
    }
#pragma unroll
    for (int i = 0; i < 4; ++i) o[i] = zf;

    const int NT = SEQ / 64;
    for (int kt = 0; kt < NT; ++kt) {
        const int key0 = kt * 64;
        const int buf  = kt & 1;
        __syncthreads();  // all waves done reading the buffer we are about to refill
        if (w == 0) {
            if (kt + 1 < NT) {
                const int nb = (kt + 1) & 1;
                tdm_load_2d((unsigned)(size_t)(void*)&sK[nb][0],
                            khb + (size_t)(key0 + 64) * HDIM, 64, 64, HDIM);
                tdm_load_2d((unsigned)(size_t)(void*)&sVT[nb][0],
                            vtb + (key0 + 64), 64, 64, SEQ);
                __builtin_amdgcn_s_wait_tensorcnt(2);  // current pair landed
            } else {
                __builtin_amdgcn_s_wait_tensorcnt(0);
            }
        }
        __syncthreads();  // current tiles visible to all waves
        const _Float16* curK = &sK[buf][0];
        const _Float16* curV = &sVT[buf][0];

        // S = Q K^T (4 column tiles of 16 keys)
        v8f s[4];
#pragma unroll
        for (int c = 0; c < 4; ++c) {
            s[c] = zf;
#pragma unroll
            for (int ks = 0; ks < 2; ++ks) {
                const _Float16* p = curK + (c * 16 + r15) * 64 + ks * 32 + h8;
                v16h bK = frag_from2(p, p + 16);
                s[c] = __builtin_amdgcn_wmma_f32_16x16x32_f16(
                    false, aQ[ks], false, bK, (short)0, s[c], false, false);
            }
        }

        // relative-key additive term
        const int cls = (key0 + 128 <= qb) ? 0 : ((key0 >= qb + 128) ? 2 : 1);
        if (cls == 0) {
#pragma unroll
            for (int c = 0; c < 4; ++c)
#pragma unroll
                for (int v = 0; v < 8; ++v) s[c][v] += srl[v][0];
        } else if (cls == 2) {
#pragma unroll
            for (int c = 0; c < 4; ++c)
#pragma unroll
                for (int v = 0; v < 8; ++v) s[c][v] += srl[v][4];
        } else {
#pragma unroll
            for (int c = 0; c < 4; ++c) {
                const int k = key0 + c * 16 + r15;
#pragma unroll
                for (int v = 0; v < 8; ++v) {
                    const int dist = k - (qb + w * 16 + v + h8);
                    float sv = srl[v][0];
                    sv = (dist >= -1) ? srl[v][1] : sv;
                    sv = (dist >= 0)  ? srl[v][2] : sv;
                    sv = (dist >= 1)  ? srl[v][3] : sv;
                    sv = (dist >= 2)  ? srl[v][4] : sv;
                    s[c][v] += sv;
                }
            }
        }

        // online softmax update
        float fac[8];
#pragma unroll
        for (int v = 0; v < 8; ++v) {
            float mx = fmaxf(fmaxf(s[0][v], s[1][v]), fmaxf(s[2][v], s[3][v]));
            mx = red_max16(mx);
            const float nm = fmaxf(m[v], mx);
            fac[v] = __expf(m[v] - nm);
            m[v]   = nm;
            l[v]  *= fac[v];
            g1[v] *= fac[v]; g2[v] *= fac[v]; g3[v] *= fac[v]; g4[v] *= fac[v];
        }
#pragma unroll
        for (int c = 0; c < 4; ++c)
#pragma unroll
            for (int v = 0; v < 8; ++v) o[c][v] *= fac[v];

        // p = exp(s-m): stats + stage to LDS (C-layout -> A-layout routing)
#pragma unroll
        for (int c = 0; c < 4; ++c) {
            const int k = key0 + c * 16 + r15;
#pragma unroll
            for (int v = 0; v < 8; ++v) {
                const float p = __expf(s[c][v] - m[v]);
                l[v] += p;
                if (cls == 2) {
                    g4[v] += p;
                } else if (cls == 1) {
                    const int dist = k - (qb + w * 16 + v + h8);
                    g1[v] += (dist == -1) ? p : 0.f;
                    g2[v] += (dist == 0)  ? p : 0.f;
                    g3[v] += (dist == 1)  ? p : 0.f;
                    g4[v] += (dist >= 2)  ? p : 0.f;
                }
                sP[w * 1024 + (v + h8) * 64 + c * 16 + r15] = (_Float16)p;
            }
        }
        // wave-private LDS RAW: DS ops are in-order per wave; wait them out.
        asm volatile("s_wait_dscnt 0x0" ::: "memory");

        // O += P * V
#pragma unroll
        for (int nt = 0; nt < 4; ++nt) {
#pragma unroll
            for (int ks = 0; ks < 2; ++ks) {
                const _Float16* ap = sP + w * 1024 + r15 * 64 + ks * 32 + h8;
                v16h aP = frag_from2(ap, ap + 16);
                const _Float16* bp = curV + (nt * 16 + r15) * 64 + ks * 32 + h8;
                v16h bV = frag_from2(bp, bp + 16);
                o[nt] = __builtin_amdgcn_wmma_f32_16x16x32_f16(
                    false, aP, false, bV, (short)0, o[nt], false, false);
            }
        }
    }

    // finalize: reduce row stats, add relative-value contribution, normalize
    float G0[8], G1[8], G2[8], G3[8], G4[8], inv[8];
#pragma unroll
    for (int v = 0; v < 8; ++v) {
        const float lr = red_sum16(l[v]);
        G1[v] = red_sum16(g1[v]);
        G2[v] = red_sum16(g2[v]);
        G3[v] = red_sum16(g3[v]);
        G4[v] = red_sum16(g4[v]);
        G0[v] = lr - G1[v] - G2[v] - G3[v] - G4[v];
        inv[v] = 1.f / lr;
    }
#pragma unroll
    for (int nt = 0; nt < 4; ++nt) {
        const int d  = nt * 16 + r15;
        const float r0 = sRel[d], r1 = sRel[64 + d], r2 = sRel[128 + d];
        const float r3 = sRel[192 + d], r4 = sRel[256 + d];
#pragma unroll
        for (int v = 0; v < 8; ++v) {
            const int row = qb + w * 16 + v + h8;
            const float val = (o[nt][v] + G0[v] * r0 + G1[v] * r1 + G2[v] * r2 +
                               G3[v] * r3 + G4[v] * r4) * inv[v];
            ctx[((size_t)(blockIdx.z * SEQ + row)) * HID + blockIdx.y * HDIM + d] =
                (_Float16)val;
        }
    }
}

// ---------------------------------------------------------------------------
extern "C" void kernel_launch(void* const* d_in, const int* in_sizes, int n_in,
                              void* d_out, int out_size, void* d_ws, size_t ws_size,
                              hipStream_t stream) {
    (void)in_sizes; (void)n_in; (void)out_size; (void)ws_size;
    const float* query = (const float*)d_in[0];
    const float* key_  = (const float*)d_in[1];
    const float* value = (const float*)d_in[2];
    const float* Wq = (const float*)d_in[3];
    const float* bq = (const float*)d_in[4];
    const float* Wk = (const float*)d_in[5];
    const float* bk = (const float*)d_in[6];
    const float* Wv = (const float*)d_in[7];
    const float* bv = (const float*)d_in[8];
    const float* Wo = (const float*)d_in[9];
    const float* bo = (const float*)d_in[10];
    const float* rel_k = (const float*)d_in[11];
    const float* rel_v = (const float*)d_in[12];

    char* ws = (char*)d_ws;
    _Float16* WqT = (_Float16*)(ws + ((size_t)0 << 20));
    _Float16* WkT = (_Float16*)(ws + ((size_t)2 << 20));
    _Float16* WvT = (_Float16*)(ws + ((size_t)4 << 20));
    _Float16* WoT = (_Float16*)(ws + ((size_t)6 << 20));
    _Float16* qh  = (_Float16*)(ws + ((size_t)8 << 20));   // [B,H,L,D], scaled 1/8
    _Float16* kh  = (_Float16*)(ws + ((size_t)16 << 20));  // [B,H,L,D]
    _Float16* vhT = (_Float16*)(ws + ((size_t)24 << 20));  // [B,H,D,L]
    _Float16* ctx = (_Float16*)(ws + ((size_t)32 << 20));  // [B,L,HID]
    float*    srl = (float*)   (ws + ((size_t)40 << 20));  // [B*H*L][5]

    // 1) weight transpose-convert
    k_wT<<<dim3(16, 16, 4), 256, 0, stream>>>(Wq, Wk, Wv, Wo, WqT, WkT, WvT, WoT);
    // 2) projections (WMMA GEMMs, TDM-staged weights)
    k_gemm<0, 0><<<dim3(8, 64), 256, 0, stream>>>((const void*)query, WqT, bq,
                                                  (void*)qh, 0.125f);
    k_gemm<0, 0><<<dim3(8, 64), 256, 0, stream>>>((const void*)key_, WkT, bk,
                                                  (void*)kh, 1.0f);
    k_gemm<0, 1><<<dim3(8, 64), 256, 0, stream>>>((const void*)value, WvT, bv,
                                                  (void*)vhT, 1.0f);
    // 3) relative-key scores
    k_srel<<<dim3((BATCH * HEADS * SEQ * 5 + 255) / 256), 256, 0, stream>>>(qh, rel_k, srl);
    // 4) flash attention + relative values (TDM double-buffered K/V^T)
    k_attn<<<dim3(SEQ / 64, HEADS, BATCH), 128, 0, stream>>>(qh, kh, vhT, srl, rel_v, ctx);
    // 5) output projection -> fp32 d_out
    k_gemm<1, 2><<<dim3(8, 64), 256, 0, stream>>>((const void*)ctx, WoT, bo,
                                                  d_out, 1.0f);
}
